// PCNetwork_65730179498621
// MI455X (gfx1250) — matrix operand order repaired and checked by reading, not compiled, not run on hardware
//
#include <hip/hip_runtime.h>

// ---------------------------------------------------------------------------
// CDNA5 (gfx1250) fused predictive-coding settle.
// Sizes: [512, 256, 256, 128], K=16 iterations, eta=0.1, clip=1.0, B=16384.
// Each workgroup owns 16 batch rows, keeps all latent state in LDS for the
// entire 16-iteration settle, and does all 6 GEMMs per iteration with
// v_wmma_f32_16x16x32_bf16. Weights are converted once to bf16 (both W and
// W^T layouts) so B-fragments are contiguous per-lane loads from L2.
// This round: branch-free hardware tanh (v_exp_f32 + v_rcp_f32) and paired
// output tiles per A-fragment load (half the LDS traffic, two independent
// WMMA accumulator chains).
// ---------------------------------------------------------------------------

typedef __attribute__((ext_vector_type(16))) __bf16        v16bf;
typedef __attribute__((ext_vector_type(8)))  float         v8f;
typedef __attribute__((ext_vector_type(4)))  unsigned int  u32x4;

#define D0 512
#define D1 256
#define D2 256
#define D3 128
#define ROWS 16
#define NTHREADS 256
#define KITERS 16

static __device__ __forceinline__ unsigned short f2bf(float f) {
  unsigned int u = __builtin_bit_cast(unsigned int, f);
  u += 0x7FFFu + ((u >> 16) & 1u);           // round-to-nearest-even
  return (unsigned short)(u >> 16);
}
static __device__ __forceinline__ float bf2f(unsigned short h) {
  unsigned int u = ((unsigned int)h) << 16;
  return __builtin_bit_cast(float, u);
}

// Branch-free tanh on the hardware transcendental pipe:
//   tanh(x) = 1 - 2/(exp(2x)+1) ;  exp(2x) = exp2(x * 2*log2(e))
// Saturates correctly for large |x| (exp2 -> inf/0, rcp -> 0/1).
static __device__ __forceinline__ float tanh_fast(float x) {
  float e = __builtin_amdgcn_exp2f(x * 2.885390081777926814f);
  return 1.0f - 2.0f * __builtin_amdgcn_rcpf(e + 1.0f);
}

union Frag { u32x4 q[2]; v16bf v; };

// Two 16x16 C-tiles of A(16xKT) * B(KT x 16) sharing one A-fragment load.
//  A: LDS, row-major bf16, row stride sA elements.
//  B: global bf16; GEMM-B element [k][n] lives at B[n*ldk + k]
//     (contiguous along k per output column -> per-lane two contiguous 16B loads).
template <int KT>
static __device__ __forceinline__ void tile_gemm_pair(
    const unsigned short* __restrict__ A, int sA,
    const unsigned short* __restrict__ B, int ldk,
    int nt0, int nt1, int lane, v8f& acc0, v8f& acc1)
{
  const int rowA = lane & 15;
  const int half = lane >> 4;                 // 0 or 1
  // A layout (16-bit A 16x32): lanes 0-15 hold K 0..7 / 16..23,
  //                            lanes 16-31 hold K 8..15 / 24..31.
  const unsigned short* abase = A + rowA * sA + half * 8;
  // B layout (16-bit B 32x16): lanes 0-15 hold K 0..7 / 16..23 of column (lane&15),
  //                            lanes 16-31 hold K 8..15 / 24..31.
  const unsigned short* b0 = B + (size_t)(nt0 * 16 + (lane & 15)) * ldk + half * 8;
  const unsigned short* b1 = B + (size_t)(nt1 * 16 + (lane & 15)) * ldk + half * 8;
#pragma unroll 4
  for (int kb = 0; kb < KT; kb += 32) {
    Frag af, bf0, bf1;
    const u32x4* ap = (const u32x4*)(abase + kb);
    af.q[0] = ap[0];          // K offs +0..7  (within half)
    af.q[1] = ap[2];          // K offs +16..23
    const u32x4* p0 = (const u32x4*)(b0 + kb);
    bf0.q[0] = p0[0]; bf0.q[1] = p0[2];
    const u32x4* p1 = (const u32x4*)(b1 + kb);
    bf1.q[0] = p1[0]; bf1.q[1] = p1[2];
    acc0 = __builtin_amdgcn_wmma_f32_16x16x32_bf16(
        false, af.v, false, bf0.v, (short)0, acc0, false, false);
    acc1 = __builtin_amdgcn_wmma_f32_16x16x32_bf16(
        false, af.v, false, bf1.v, (short)0, acc1, false, false);
  }
}

// Single-tile variant (used where a wave owns an odd tile count).
template <int KT>
static __device__ __forceinline__ v8f tile_gemm(
    const unsigned short* __restrict__ A, int sA,
    const unsigned short* __restrict__ B, int ldk,
    int nt, int lane, v8f acc)
{
  const int rowA = lane & 15;
  const int half = lane >> 4;
  const unsigned short* abase = A + rowA * sA + half * 8;
  const unsigned short* bbase = B + (size_t)(nt * 16 + (lane & 15)) * ldk + half * 8;
#pragma unroll 4
  for (int kb = 0; kb < KT; kb += 32) {
    Frag af, bf;
    const u32x4* ap = (const u32x4*)(abase + kb);
    af.q[0] = ap[0];
    af.q[1] = ap[2];
    const u32x4* bp = (const u32x4*)(bbase + kb);
    bf.q[0] = bp[0];
    bf.q[1] = bp[2];
    acc = __builtin_amdgcn_wmma_f32_16x16x32_bf16(
        false, af.v, false, bf.v, (short)0, acc, false, false);
  }
  return acc;
}

// Convert f32 weight (rows x cols, row-major) into bf16 copies: same layout
// (Wb) and transposed layout (WT, cols x rows row-major).
__global__ void prep_w(const float* __restrict__ W, int rows, int cols,
                       unsigned short* __restrict__ Wb,
                       unsigned short* __restrict__ WT)
{
  int idx = blockIdx.x * blockDim.x + threadIdx.x;
  int n = rows * cols;
  if (idx >= n) return;
  int r = idx / cols;
  int c = idx - r * cols;
  unsigned short h = f2bf(W[idx]);
  Wb[idx] = h;
  WT[(size_t)c * rows + r] = h;
}

// ---- LDS layout (bytes), total 159744 (~156 KB, fits 320 KB/WGP) ----------
#define OFF_X0S   0        // f32 16x512 = 32768
#define OFF_X1    32768    // f32 16x256 = 16384
#define OFF_X2    49152    // f32 16x256 = 16384
#define OFF_X3    65536    // f32 16x128 =  8192
#define OFF_XB1   73728    // bf16 16x256 = 8192
#define OFF_XB2   81920    // bf16 16x256 = 8192
#define OFF_XB3   90112    // bf16 16x128 = 4096
#define OFF_T0    94208    // bf16 16x512 = 16384
#define OFF_T1    110592   // bf16 16x256 = 8192
#define OFF_T2    118784   // bf16 16x256 = 8192
#define OFF_G0    126976   // bf16 16x512 = 16384
#define OFF_G1    143360   // bf16 16x256 = 8192
#define OFF_G2    151552   // bf16 16x256 = 8192
#define LDS_BYTES 159744

__global__ void __launch_bounds__(NTHREADS)
settle_kernel(const float* __restrict__ x0,
              const unsigned short* __restrict__ Wb0,
              const unsigned short* __restrict__ WT0,
              const unsigned short* __restrict__ Wb1,
              const unsigned short* __restrict__ WT1,
              const unsigned short* __restrict__ Wb2,
              const unsigned short* __restrict__ WT2,
              float* __restrict__ out)
{
  extern __shared__ char smem[];
  float*          x0s = (float*)(smem + OFF_X0S);
  float*          x1  = (float*)(smem + OFF_X1);
  float*          x2  = (float*)(smem + OFF_X2);
  float*          x3  = (float*)(smem + OFF_X3);
  unsigned short* xb1 = (unsigned short*)(smem + OFF_XB1);
  unsigned short* xb2 = (unsigned short*)(smem + OFF_XB2);
  unsigned short* xb3 = (unsigned short*)(smem + OFF_XB3);
  unsigned short* t0  = (unsigned short*)(smem + OFF_T0);
  unsigned short* t1  = (unsigned short*)(smem + OFF_T1);
  unsigned short* t2  = (unsigned short*)(smem + OFF_T2);
  unsigned short* g0  = (unsigned short*)(smem + OFF_G0);
  unsigned short* g1  = (unsigned short*)(smem + OFF_G1);
  unsigned short* g2  = (unsigned short*)(smem + OFF_G2);

  const int tid  = threadIdx.x;
  const int lane = tid & 31;           // wave32
  const int wv   = tid >> 5;           // 8 waves per workgroup
  const size_t rowBase = (size_t)blockIdx.x * ROWS;

  // Load clamped input tile (contiguous) and zero latent state.
  for (int i = tid; i < ROWS * D0; i += NTHREADS) x0s[i] = x0[rowBase * D0 + i];
  for (int i = tid; i < ROWS * D1; i += NTHREADS) { x1[i] = 0.f; xb1[i] = 0; }
  for (int i = tid; i < ROWS * D2; i += NTHREADS) { x2[i] = 0.f; xb2[i] = 0; }
  for (int i = tid; i < ROWS * D3; i += NTHREADS) { x3[i] = 0.f; xb3[i] = 0; }
  __syncthreads();

  const int colL = lane & 15;
  const int mb   = (lane >> 4) * 8;    // C/D: lanes 0-15 -> M 0..7, 16-31 -> M 8..15

#define ZACC {0.f,0.f,0.f,0.f,0.f,0.f,0.f,0.f}
#define STORE_TANH(TP, STRIDE, NT, ACC)                                      \
  { int c_ = (NT) * 16 + colL;                                               \
    _Pragma("unroll")                                                        \
    for (int r_ = 0; r_ < 8; ++r_)                                           \
      (TP)[(mb + r_) * (STRIDE) + c_] = f2bf(tanh_fast((ACC)[r_])); }

  for (int it = 0; it < KITERS; ++it) {
    // ---- forward predictions: t_l = tanh(x_{l+1} @ W_l) ----
    {                                                     // pre0: K=D1, N=D0 (32 tiles)
      v8f a0 = ZACC, a1 = ZACC;
      tile_gemm_pair<D1>(xb1, D1, WT0, D1, wv, wv + 8, lane, a0, a1);
      STORE_TANH(t0, D0, wv, a0);
      STORE_TANH(t0, D0, wv + 8, a1);
      v8f a2 = ZACC, a3 = ZACC;
      tile_gemm_pair<D1>(xb1, D1, WT0, D1, wv + 16, wv + 24, lane, a2, a3);
      STORE_TANH(t0, D0, wv + 16, a2);
      STORE_TANH(t0, D0, wv + 24, a3);
    }
    {                                                     // pre1: K=D2, N=D1 (16 tiles)
      v8f a0 = ZACC, a1 = ZACC;
      tile_gemm_pair<D2>(xb2, D2, WT1, D2, wv, wv + 8, lane, a0, a1);
      STORE_TANH(t1, D1, wv, a0);
      STORE_TANH(t1, D1, wv + 8, a1);
    }
    {                                                     // pre2: K=D3, N=D2 (16 tiles)
      v8f a0 = ZACC, a1 = ZACC;
      tile_gemm_pair<D3>(xb3, D3, WT2, D3, wv, wv + 8, lane, a0, a1);
      STORE_TANH(t2, D2, wv, a0);
      STORE_TANH(t2, D2, wv + 8, a1);
    }
    __syncthreads();

    // ---- g_l = (x_l - t_l) * (1 - t_l^2)  (all from pre-update state) ----
    for (int i = tid; i < ROWS * D0; i += NTHREADS) {
      float t = bf2f(t0[i]); g0[i] = f2bf((x0s[i] - t) * (1.f - t * t));
    }
    for (int i = tid; i < ROWS * D1; i += NTHREADS) {
      float t = bf2f(t1[i]); g1[i] = f2bf((x1[i] - t) * (1.f - t * t));
    }
    for (int i = tid; i < ROWS * D2; i += NTHREADS) {
      float t = bf2f(t2[i]); g2[i] = f2bf((x2[i] - t) * (1.f - t * t));
    }
    __syncthreads();

    // ---- feedback + update (passes touch disjoint state; no inner barrier)
    {                                                     // dx1 = g0 @ W0^T - e1
      v8f a0 = ZACC, a1 = ZACC;
      tile_gemm_pair<D0>(g0, D0, Wb0, D0, wv, wv + 8, lane, a0, a1);
#pragma unroll
      for (int p = 0; p < 2; ++p) {
        v8f& a = p ? a1 : a0;
        int c = (wv + 8 * p) * 16 + colL;
#pragma unroll
        for (int r = 0; r < 8; ++r) {
          int idx = (mb + r) * D1 + c;
          float dx = a[r] - (x1[idx] - bf2f(t1[idx]));
          dx = fminf(fmaxf(dx, -1.f), 1.f);
          float xn = x1[idx] + 0.1f * dx;
          x1[idx] = xn; xb1[idx] = f2bf(xn);
        }
      }
    }
    {                                                     // dx2 = g1 @ W1^T - e2
      v8f a0 = ZACC, a1 = ZACC;
      tile_gemm_pair<D1>(g1, D1, Wb1, D1, wv, wv + 8, lane, a0, a1);
#pragma unroll
      for (int p = 0; p < 2; ++p) {
        v8f& a = p ? a1 : a0;
        int c = (wv + 8 * p) * 16 + colL;
#pragma unroll
        for (int r = 0; r < 8; ++r) {
          int idx = (mb + r) * D2 + c;
          float dx = a[r] - (x2[idx] - bf2f(t2[idx]));
          dx = fminf(fmaxf(dx, -1.f), 1.f);
          float xn = x2[idx] + 0.1f * dx;
          x2[idx] = xn; xb2[idx] = f2bf(xn);
        }
      }
    }
    {                                                     // dx3 = g2 @ W2^T (8 tiles)
      v8f a0 = ZACC;
      a0 = tile_gemm<D2>(g2, D2, Wb2, D2, wv, lane, a0);
      int c = wv * 16 + colL;
#pragma unroll
      for (int r = 0; r < 8; ++r) {
        int idx = (mb + r) * D3 + c;
        float dx = fminf(fmaxf(a0[r], -1.f), 1.f);
        float xn = x3[idx] + 0.1f * dx;
        x3[idx] = xn; xb3[idx] = f2bf(xn);
      }
    }
    __syncthreads();
  }

  // settled top-layer latent -> output (contiguous tile)
  for (int i = tid; i < ROWS * D3; i += NTHREADS) out[rowBase * D3 + i] = x3[i];
}

// ---------------------------------------------------------------------------
extern "C" void kernel_launch(void* const* d_in, const int* in_sizes, int n_in,
                              void* d_out, int out_size, void* d_ws, size_t ws_size,
                              hipStream_t stream) {
  const float* x0 = (const float*)d_in[0];   // (B, 512)
  const float* W0 = (const float*)d_in[1];   // (256, 512)
  const float* W1 = (const float*)d_in[2];   // (256, 256)
  const float* W2 = (const float*)d_in[3];   // (128, 256)

  // bf16 weight copies in workspace (element offsets):
  unsigned short* ws  = (unsigned short*)d_ws;
  unsigned short* Wb0 = ws;                         // 256*512
  unsigned short* WT0 = ws + 131072;                // 512*256
  unsigned short* Wb1 = ws + 262144;                // 256*256
  unsigned short* WT1 = ws + 327680;                // 256*256
  unsigned short* Wb2 = ws + 393216;                // 128*256
  unsigned short* WT2 = ws + 425984;                // 256*128   (end 458752 elems = 896 KB)

  prep_w<<<(256 * 512 + 255) / 256, 256, 0, stream>>>(W0, 256, 512, Wb0, WT0);
  prep_w<<<(256 * 256 + 255) / 256, 256, 0, stream>>>(W1, 256, 256, Wb1, WT1);
  prep_w<<<(128 * 256 + 255) / 256, 256, 0, stream>>>(W2, 128, 256, Wb2, WT2);

  const int Bn   = in_sizes[0] / D0;       // 16384
  const int nblk = Bn / ROWS;              // 1024 workgroups
  settle_kernel<<<nblk, NTHREADS, LDS_BYTES, stream>>>(
      x0, Wb0, WT0, Wb1, WT1, Wb2, WT2, (float*)d_out);
}
